// SGN_10299331576369
// MI455X (gfx1250) — matrix operand order repaired and checked by reading, not compiled, not run on hardware
//
#include <hip/hip_runtime.h>

#define DEVINL __device__ __forceinline__

typedef __attribute__((ext_vector_type(16))) __bf16 v16bf;
typedef __attribute__((ext_vector_type(8)))  __bf16 v8bf;
typedef __attribute__((ext_vector_type(4)))  __bf16 v4bf;
typedef __attribute__((ext_vector_type(8)))  float  v8f;
typedef __attribute__((ext_vector_type(4)))  float  f32x4;

constexpr float kEPS = 1e-5f;
constexpr float kNEG = -3.402823466e38f;

// ---------------- WMMA helpers (CDNA5 16x16x32 bf16, f32 accumulate) ----------

DEVINL v8f vzero() { v8f z = {0.f,0.f,0.f,0.f,0.f,0.f,0.f,0.f}; return z; }

DEVINL v8f wmma_bf16(v16bf a, v16bf b, v8f c) {
  // 8 args: (neg_a, A, neg_b, B, c_mod, C, reuse_a, reuse_b)
  return __builtin_amdgcn_wmma_f32_16x16x32_bf16(false, a, false, b, (short)0, c,
                                                 false, false);
}

DEVINL v16bf cat8(v8bf lo, v8bf hi) {
  return __builtin_shufflevector(lo, hi, 0,1,2,3,4,5,6,7,8,9,10,11,12,13,14,15);
}

// A 16x32 bf16 from row-major LDS src[row][k]: lane L reads row (L&15),
// halves K = k0+off8+{0..7} and k0+off8+{16..23}  ->  2x ds_load_b128.
DEVINL v16bf ldA_lds(const __bf16* s, int pitch, int row0, int k0, int lane) {
  const __bf16* r = s + (row0 + (lane & 15)) * pitch + k0 + ((lane & 16) ? 8 : 0);
  v8bf lo = *(const v8bf*)r;
  v8bf hi = *(const v8bf*)(r + 16);
  return cat8(lo, hi);
}
// B 32x16 bf16 from N-major (transposed) LDS srcT[n][k]: lane reads
// halves K = k0+off8+{0..7} and k0+off8+{16..23}  ->  2x ds_load_b128.
DEVINL v16bf ldB_ldsT(const __bf16* s, int pitch, int k0, int n0, int lane) {
  const __bf16* r = s + (n0 + (lane & 15)) * pitch + k0 + ((lane & 16) ? 8 : 0);
  v8bf lo = *(const v8bf*)r;
  v8bf hi = *(const v8bf*)(r + 16);
  return cat8(lo, hi);
}
// B 32x16 from N-major bf16 GLOBAL srcT[n][k] (softmaxed G) -> 2x b128 loads.
DEVINL v16bf ldB_bf16gT(const __bf16* s, int pitch, int k0, int n0, int lane) {
  const __bf16* r = s + (size_t)(n0 + (lane & 15)) * pitch + k0 + ((lane & 16) ? 8 : 0);
  v8bf lo = *(const v8bf*)r;
  v8bf hi = *(const v8bf*)(r + 16);
  return cat8(lo, hi);
}
// A 16x32 gathered from row-major f32 GLOBAL (L2-resident weights): 4x b128.
DEVINL v16bf ldA_f32g(const float* s, int pitch, int row0, int k0, int lane) {
  const float* r = s + (size_t)(row0 + (lane & 15)) * pitch + k0 + ((lane & 16) ? 8 : 0);
  f32x4 a0 = *(const f32x4*)r;
  f32x4 a1 = *(const f32x4*)(r + 4);
  f32x4 b0 = *(const f32x4*)(r + 16);
  f32x4 b1 = *(const f32x4*)(r + 20);
  v16bf a;
#pragma unroll
  for (int i = 0; i < 4; ++i) {
    a[i]      = (__bf16)a0[i];
    a[4 + i]  = (__bf16)a1[i];
    a[8 + i]  = (__bf16)b0[i];
    a[12 + i] = (__bf16)b1[i];
  }
  return a;
}

// ---------------- 1) spa1/tem1 tables (one-hot inputs collapse layer 1) -------

__global__ void k_embed(const float* __restrict__ se1_w, const float* __restrict__ se1_b,
                        const float* __restrict__ se2_w, const float* __restrict__ se2_b,
                        const float* __restrict__ te1_w, const float* __restrict__ te1_b,
                        const float* __restrict__ te2_w, const float* __restrict__ te2_b,
                        float* __restrict__ spa_tab, float* __restrict__ tem_tab) {
  int i = blockIdx.x;  // j or t in 0..223
  int tid = threadIdx.x;
  __shared__ float sh[64], th[64];
  if (tid < 64) {
    sh[tid] = fmaxf(se1_w[tid * 224 + i] + se1_b[tid], 0.f);
    th[tid] = fmaxf(te1_w[tid * 224 + i] + te1_b[tid], 0.f);
  }
  __syncthreads();
  if (tid < 64) {
    float a = se2_b[tid];
    for (int c = 0; c < 64; ++c) a += se2_w[tid * 64 + c] * sh[c];
    spa_tab[tid * 224 + i] = fmaxf(a, 0.f);
  }
  {
    float a = te2_b[tid];
    for (int c = 0; c < 64; ++c) a += te2_w[tid * 64 + c] * th[c];
    tem_tab[tid * 224 + i] = fmaxf(a, 0.f);
  }
}

// ---------------- 2) bn0 + joint embed + concat -> H1 slab layout -------------

__global__ void k_build_h(const float* __restrict__ x, const float* __restrict__ bn0,
                          const float* __restrict__ je1_w, const float* __restrict__ je1_b,
                          const float* __restrict__ je2_w, const float* __restrict__ je2_b,
                          const float* __restrict__ spa, float* __restrict__ H1) {
  int s = blockIdx.x, b = s / 224, t = s % 224;
  int tid = threadIdx.x;
  __shared__ float w1s[192], b1s[64], w2s[4096], b2s[64];
  for (int i = tid; i < 192; i += 256) w1s[i] = je1_w[i];
  for (int i = tid; i < 4096; i += 256) w2s[i] = je2_w[i];
  if (tid < 64) { b1s[tid] = je1_b[tid]; b2s[tid] = je2_b[tid]; }
  __syncthreads();
  int j = tid;
  if (j < 224) {
    float xf[3];
#pragma unroll
    for (int c = 0; c < 3; ++c) {
      int ch = c * 224 + j;  // bn0 channel = c*J + j after reshape
      float xv = x[(((size_t)b * 3 + c) * 224 + j) * 224 + t];
      float inv = bn0[ch] / sqrtf(bn0[2016 + ch] + kEPS);
      xf[c] = (xv - bn0[1344 + ch]) * inv + bn0[672 + ch];
    }
    float v[64];
#pragma unroll
    for (int o = 0; o < 64; ++o)
      v[o] = fmaxf(w1s[o * 3] * xf[0] + w1s[o * 3 + 1] * xf[1] + w1s[o * 3 + 2] * xf[2] + b1s[o], 0.f);
    float* dst = H1 + (size_t)s * 128 * 224;
    for (int o = 0; o < 64; ++o) {
      float a = b2s[o];
#pragma unroll
      for (int c = 0; c < 64; ++c) a += w2s[o * 64 + c] * v[c];
      dst[o * 224 + j] = fmaxf(a, 0.f);            // pos channels 0..63
      dst[(64 + o) * 224 + j] = spa[o * 224 + j];  // spa1 channels 64..127
    }
  }
}

// ---------------- 3) fused Q,K build + attention scores (WMMA, slab in LDS) ---
// LDS: HsT[j][c] 224x128, QsT[j][c] 224x256, KsT[k][c] 224x256  (286720 B)

__global__ void __launch_bounds__(256) k_attn(
    const float* __restrict__ H1, const float* __restrict__ g1_w, const float* __restrict__ g1_b,
    const float* __restrict__ g2_w, const float* __restrict__ g2_b, float* __restrict__ G) {
  extern __shared__ char smem[];
  __bf16* HsT = (__bf16*)smem;                                    // 224 x 128
  __bf16* QsT = (__bf16*)(smem + 224 * 128 * 2);                  // 224 x 256
  __bf16* KsT = (__bf16*)(smem + 224 * 128 * 2 + 224 * 256 * 2);  // 224 x 256
  int s = blockIdx.x;
  int tid = threadIdx.x, lane = tid & 31, wave = tid >> 5;
  int hi = (lane >> 4) & 1;

  // stage H transposed: coalesced float4 reads, 4 scattered b16 LDS writes
  const float* hsrc = H1 + (size_t)s * 128 * 224;
  for (int q = tid; q < 128 * 224 / 4; q += 256) {
    f32x4 v4 = *(const f32x4*)(hsrc + q * 4);
    int c = (q * 4) / 224, j = (q * 4) % 224;
#pragma unroll
    for (int e = 0; e < 4; ++e) HsT[(j + e) * 128 + c] = (__bf16)v4[e];
  }
  __syncthreads();

  // Phase 1: waves 0-3 build Q = g1_w*H + g1_b; waves 4-7 build K.
  const float* Wg = (wave < 4) ? g1_w : g2_w;
  const float* Bg = (wave < 4) ? g1_b : g2_b;
  __bf16* Dst = (wave < 4) ? QsT : KsT;
  int gw = wave & 3;
  for (int jc = 0; jc < 2; ++jc)
    for (int mp = 0; mp < 2; ++mp) {
      int mt = mp * 8 + gw * 2;  // 2 M-tiles (32 out channels) per pass
      v8f acc[2][7];
#pragma unroll
      for (int mi = 0; mi < 2; ++mi)
#pragma unroll
        for (int n = 0; n < 7; ++n) acc[mi][n] = vzero();
      for (int kb = 0; kb < 4; ++kb) {  // K = 128
        v16bf A0 = ldA_f32g(Wg, 128, (mt + 0) * 16, kb * 32, lane);
        v16bf A1 = ldA_f32g(Wg, 128, (mt + 1) * 16, kb * 32, lane);
#pragma unroll
        for (int n = 0; n < 7; ++n) {
          v16bf Bf = ldB_ldsT(HsT, 128, kb * 32, jc * 112 + n * 16, lane);
          acc[0][n] = wmma_bf16(A0, Bf, acc[0][n]);
          acc[1][n] = wmma_bf16(A1, Bf, acc[1][n]);
        }
      }
#pragma unroll
      for (int mi = 0; mi < 2; ++mi)
#pragma unroll
        for (int r = 0; r < 8; ++r) {
          int o = (mt + mi) * 16 + r + hi * 8;
          float bs = Bg[o];
#pragma unroll
          for (int n = 0; n < 7; ++n) {
            int j = jc * 112 + n * 16 + (lane & 15);
            Dst[j * 256 + o] = (__bf16)(acc[mi][n][r] + bs);  // transposed store
          }
        }
    }
  __syncthreads();

  // Phase 2: S[j,k] = sum_c Q[c,j] K[c,k]. QsT is row-major in (j,c) => A.
  float* gdst = G + (size_t)s * 224 * 224;
  for (int m = wave; m < 14; m += 8) {
    v8f acc[14];
#pragma unroll
    for (int n = 0; n < 14; ++n) acc[n] = vzero();
    for (int kb = 0; kb < 8; ++kb) {
      v16bf Af = ldA_lds(QsT, 256, m * 16, kb * 32, lane);
#pragma unroll
      for (int n = 0; n < 14; ++n) {
        v16bf Bf = ldB_ldsT(KsT, 256, kb * 32, n * 16, lane);
        acc[n] = wmma_bf16(Af, Bf, acc[n]);
      }
    }
#pragma unroll
    for (int n = 0; n < 14; ++n)
#pragma unroll
      for (int r = 0; r < 8; ++r) {
        int j = m * 16 + r + hi * 8;
        int k = n * 16 + (lane & 15);
        gdst[(size_t)j * 224 + k] = acc[n][r];
      }
  }
}

// ---------------- 4) row softmax over k; emit bf16 copy for the GCN B operand -

__global__ void k_softmax(const float* __restrict__ G, __bf16* __restrict__ Gbf) {
  int wave = threadIdx.x >> 5, lane = threadIdx.x & 31;
  size_t row = (size_t)blockIdx.x * 8 + wave;  // 200704 rows exactly
  const float* p = G + row * 224;
  __bf16* po = Gbf + row * 224;
  float v[7];
#pragma unroll
  for (int i = 0; i < 7; ++i) v[i] = p[lane + i * 32];
  float m = v[0];
#pragma unroll
  for (int i = 1; i < 7; ++i) m = fmaxf(m, v[i]);
#pragma unroll
  for (int off = 16; off > 0; off >>= 1) m = fmaxf(m, __shfl_xor(m, off, 32));
  float sum = 0.f;
#pragma unroll
  for (int i = 0; i < 7; ++i) { v[i] = __expf(v[i] - m); sum += v[i]; }
#pragma unroll
  for (int off = 16; off > 0; off >>= 1) sum += __shfl_xor(sum, off, 32);
  float inv = 1.f / sum;
#pragma unroll
  for (int i = 0; i < 7; ++i) po[lane + i * 32] = (__bf16)(v[i] * inv);
}

// ---------------- 5) GCN layer: y = H*G^T; out = relu(bn(w*y + w1*H + b)) -----
// LDS: Hs[c][224] (phase1 A), HsT[j][c] (phase2 B), YsT[j][c] (112 rows).

template <int CIN, int COUT>
__global__ void __launch_bounds__(256) k_gcn(
    const float* __restrict__ Hin, const __bf16* __restrict__ Gbf,
    const float* __restrict__ W, const float* __restrict__ W1,
    const float* __restrict__ bias, const float* __restrict__ bn,
    float* __restrict__ Hout) {
  extern __shared__ char smem[];
  __bf16* Hs  = (__bf16*)smem;                       // CIN x 224
  __bf16* HsT = (__bf16*)(smem + CIN * 448);         // 224 x CIN
  __bf16* YsT = (__bf16*)(smem + CIN * 896);         // 112 x CIN
  int s = blockIdx.x;
  int tid = threadIdx.x, lane = tid & 31, wave = tid >> 5;
  int hi = (lane >> 4) & 1;

  const float* hsrc = Hin + (size_t)s * CIN * 224;
  for (int q = tid; q < CIN * 224 / 4; q += 256) {
    f32x4 v4 = *(const f32x4*)(hsrc + q * 4);
    int c = (q * 4) / 224, j = (q * 4) % 224;
    v4bf pk = {(__bf16)v4[0], (__bf16)v4[1], (__bf16)v4[2], (__bf16)v4[3]};
    *(v4bf*)(Hs + q * 4) = pk;  // row-major, contiguous b64 store
#pragma unroll
    for (int e = 0; e < 4; ++e) HsT[(j + e) * CIN + c] = pk[e];
  }
  __syncthreads();

  const __bf16* gsrc = Gbf + (size_t)s * 224 * 224;
  constexpr int MT1 = CIN / 128;   // M-tiles per wave in phase 1
  constexpr int MT2 = COUT / 128;  // M-tiles per wave in phase 2
  constexpr int KB2 = CIN / 32;

  for (int jc = 0; jc < 2; ++jc) {
    // Phase 1: y[c, jchunk] = sum_k H[c,k] * G[j,k]  (B = bf16 G rows from L2)
    {
      v8f acc[MT1][7];
#pragma unroll
      for (int mi = 0; mi < MT1; ++mi)
#pragma unroll
        for (int n = 0; n < 7; ++n) acc[mi][n] = vzero();
      for (int kb = 0; kb < 7; ++kb) {  // K = 224
        v16bf Af[MT1];
#pragma unroll
        for (int mi = 0; mi < MT1; ++mi)
          Af[mi] = ldA_lds(Hs, 224, (wave * MT1 + mi) * 16, kb * 32, lane);
#pragma unroll
        for (int n = 0; n < 7; ++n) {
          v16bf Bf = ldB_bf16gT(gsrc, 224, kb * 32, jc * 112 + n * 16, lane);
#pragma unroll
          for (int mi = 0; mi < MT1; ++mi)
            acc[mi][n] = wmma_bf16(Af[mi], Bf, acc[mi][n]);
        }
      }
#pragma unroll
      for (int mi = 0; mi < MT1; ++mi)
#pragma unroll
        for (int r = 0; r < 8; ++r) {
          int c = (wave * MT1 + mi) * 16 + r + hi * 8;
#pragma unroll
          for (int n = 0; n < 7; ++n)
            YsT[(n * 16 + (lane & 15)) * CIN + c] = (__bf16)acc[mi][n][r];
        }
    }
    __syncthreads();
    // Phase 2: out = W*y + W1*H + bias -> BN -> ReLU
    {
      v8f acc[MT2][7];
#pragma unroll
      for (int mi = 0; mi < MT2; ++mi)
#pragma unroll
        for (int n = 0; n < 7; ++n) acc[mi][n] = vzero();
      for (int kb = 0; kb < KB2; ++kb) {
        v16bf Aw[MT2], Aw1[MT2];
#pragma unroll
        for (int mi = 0; mi < MT2; ++mi) {
          Aw[mi]  = ldA_f32g(W,  CIN, (wave * MT2 + mi) * 16, kb * 32, lane);
          Aw1[mi] = ldA_f32g(W1, CIN, (wave * MT2 + mi) * 16, kb * 32, lane);
        }
#pragma unroll
        for (int n = 0; n < 7; ++n) {
          v16bf By = ldB_ldsT(YsT, CIN, kb * 32, n * 16, lane);
          v16bf Bh = ldB_ldsT(HsT, CIN, kb * 32, jc * 112 + n * 16, lane);
#pragma unroll
          for (int mi = 0; mi < MT2; ++mi) {
            acc[mi][n] = wmma_bf16(Aw[mi],  By, acc[mi][n]);
            acc[mi][n] = wmma_bf16(Aw1[mi], Bh, acc[mi][n]);
          }
        }
      }
      float* dst = Hout + (size_t)s * COUT * 224;
#pragma unroll
      for (int mi = 0; mi < MT2; ++mi)
#pragma unroll
        for (int r = 0; r < 8; ++r) {
          int o = (wave * MT2 + mi) * 16 + r + hi * 8;
          float inv = bn[o] / sqrtf(bn[3 * COUT + o] + kEPS);
          float sh = bias[o] - bn[2 * COUT + o];
          float bt = bn[COUT + o];
#pragma unroll
          for (int n = 0; n < 7; ++n) {
            int j = jc * 112 + n * 16 + (lane & 15);
            dst[o * 224 + j] = fmaxf((acc[mi][n][r] + sh) * inv + bt, 0.f);
          }
        }
    }
    __syncthreads();
  }
}

// ---------------- 6) + tem1, max over J -> hmax[b][c][t] ----------------------

__global__ void k_post(const float* __restrict__ H3, const float* __restrict__ tem,
                       float* __restrict__ hmax) {
  int s = blockIdx.x, b = s / 224, t = s % 224;
  int lane = threadIdx.x & 31, wave = threadIdx.x >> 5;
  for (int c = wave; c < 256; c += 8) {
    const float* r = H3 + ((size_t)s * 256 + c) * 224;
    float m = kNEG;
#pragma unroll
    for (int i = 0; i < 7; ++i) m = fmaxf(m, r[lane + i * 32]);
#pragma unroll
    for (int off = 16; off > 0; off >>= 1) m = fmaxf(m, __shfl_xor(m, off, 32));
    if (lane == 0) hmax[((size_t)b * 256 + c) * 224 + t] = m + tem[c * 224 + t];
  }
}

// ---------------- 7) adaptive pool 20 + loc1 conv3 + loc2 + global max --------

__global__ void k_final(const float* __restrict__ hmax,
                        const float* __restrict__ w1c, const float* __restrict__ b1c,
                        const float* __restrict__ bn1,
                        const float* __restrict__ w2c, const float* __restrict__ b2c,
                        const float* __restrict__ bn2, float* __restrict__ out) {
  int b = blockIdx.x, tid = threadIdx.x;
  __shared__ float pooled[256 * 20];
  __shared__ float h2[256 * 20];
  {
    int c = tid;
    if (c < 256) {
      const float* r = hmax + ((size_t)b * 256 + c) * 224;
      for (int i = 0; i < 20; ++i) {
        int s0 = (i * 224) / 20, e0 = ((i + 1) * 224 + 19) / 20;
        float m = r[s0];
        for (int q = s0 + 1; q < e0; ++q) m = fmaxf(m, r[q]);
        pooled[c * 20 + i] = m;
      }
    }
  }
  __syncthreads();
  for (int it = tid; it < 256 * 20; it += 256) {
    int o = it / 20, w = it % 20;
    float acc = b1c[o];
    for (int i = 0; i < 256; ++i) {
      const float* wp = w1c + ((size_t)o * 256 + i) * 3;
      const float* pp = pooled + i * 20;
      if (w > 0) acc += wp[0] * pp[w - 1];
      acc += wp[1] * pp[w];
      if (w < 19) acc += wp[2] * pp[w + 1];
    }
    float inv = bn1[o] / sqrtf(bn1[768 + o] + kEPS);
    h2[it] = fmaxf((acc - bn1[512 + o]) * inv + bn1[256 + o], 0.f);
  }
  __syncthreads();
  for (int o = tid; o < 512; o += 256) {
    float inv = bn2[o] / sqrtf(bn2[1536 + o] + kEPS);
    float mu = bn2[1024 + o], bt = bn2[512 + o];
    float mx = kNEG;
    for (int w = 0; w < 20; ++w) {
      float acc = b2c[o];
      for (int i = 0; i < 256; ++i) acc += w2c[(size_t)o * 256 + i] * h2[i * 20 + w];
      mx = fmaxf(mx, fmaxf((acc - mu) * inv + bt, 0.f));
    }
    out[b * 512 + o] = mx;
  }
}

// ---------------- workspace layout (floats) -----------------------------------
constexpr size_t OFF_SPA  = 0;
constexpr size_t OFF_TEM  = OFF_SPA + 64 * 224;
constexpr size_t OFF_HMAX = OFF_TEM + 256 * 224;
constexpr size_t OFF_H1   = OFF_HMAX + (size_t)4 * 256 * 224;
constexpr size_t OFF_G    = OFF_H1 + (size_t)896 * 128 * 224;
constexpr size_t OFF_A    = OFF_G + (size_t)896 * 224 * 224;
constexpr size_t OFF_B    = OFF_A + (size_t)896 * 256 * 224;
constexpr size_t OFF_GBF  = OFF_B + (size_t)896 * 256 * 224;  // bf16, 1/2 word each

extern "C" void kernel_launch(void* const* d_in, const int* in_sizes, int n_in,
                              void* d_out, int out_size, void* d_ws, size_t ws_size,
                              hipStream_t stream) {
  const float* x       = (const float*)d_in[0];
  const float* je1_w   = (const float*)d_in[1];
  const float* je1_b   = (const float*)d_in[2];
  const float* je2_w   = (const float*)d_in[3];
  const float* je2_b   = (const float*)d_in[4];
  const float* se1_w   = (const float*)d_in[5];
  const float* se1_b   = (const float*)d_in[6];
  const float* se2_w   = (const float*)d_in[7];
  const float* se2_b   = (const float*)d_in[8];
  const float* te1_w   = (const float*)d_in[9];
  const float* te1_b   = (const float*)d_in[10];
  const float* te2_w   = (const float*)d_in[11];
  const float* te2_b   = (const float*)d_in[12];
  const float* g1_w    = (const float*)d_in[13];
  const float* g1_b    = (const float*)d_in[14];
  const float* g2_w    = (const float*)d_in[15];
  const float* g2_b    = (const float*)d_in[16];
  const float* gcn1_w  = (const float*)d_in[17];
  const float* gcn1_w1 = (const float*)d_in[18];
  const float* gcn1_b  = (const float*)d_in[19];
  const float* gcn2_w  = (const float*)d_in[20];
  const float* gcn2_w1 = (const float*)d_in[21];
  const float* gcn2_b  = (const float*)d_in[22];
  const float* gcn3_w  = (const float*)d_in[23];
  const float* gcn3_w1 = (const float*)d_in[24];
  const float* gcn3_b  = (const float*)d_in[25];
  const float* loc1_w  = (const float*)d_in[26];
  const float* loc1_b  = (const float*)d_in[27];
  const float* loc2_w  = (const float*)d_in[28];
  const float* loc2_b  = (const float*)d_in[29];
  const float* bn0     = (const float*)d_in[30];
  const float* gcn1_bn = (const float*)d_in[31];
  const float* gcn2_bn = (const float*)d_in[32];
  const float* gcn3_bn = (const float*)d_in[33];
  const float* loc1_bn = (const float*)d_in[34];
  const float* loc2_bn = (const float*)d_in[35];

  float* ws = (float*)d_ws;
  float* spa  = ws + OFF_SPA;
  float* tem  = ws + OFF_TEM;
  float* hmax = ws + OFF_HMAX;
  float* H1   = ws + OFF_H1;
  float* G    = ws + OFF_G;
  float* BA   = ws + OFF_A;             // 256-ch ping
  float* BB   = ws + OFF_B;             // 256-ch pong
  __bf16* Gbf = (__bf16*)(ws + OFF_GBF);

  const int SHM_ATTN = (224 * 128 + 2 * 224 * 256) * 2;  // 286720 B
  const int SHM_G128 = 128 * 1120;                       // 143360 B
  const int SHM_G256 = 256 * 1120;                       // 286720 B
  hipFuncSetAttribute((const void*)k_attn,
                      hipFuncAttributeMaxDynamicSharedMemorySize, SHM_ATTN);
  hipFuncSetAttribute((const void*)(k_gcn<128, 128>),
                      hipFuncAttributeMaxDynamicSharedMemorySize, SHM_G128);
  hipFuncSetAttribute((const void*)(k_gcn<128, 256>),
                      hipFuncAttributeMaxDynamicSharedMemorySize, SHM_G128);
  hipFuncSetAttribute((const void*)(k_gcn<256, 256>),
                      hipFuncAttributeMaxDynamicSharedMemorySize, SHM_G256);

  k_embed<<<224, 256, 0, stream>>>(se1_w, se1_b, se2_w, se2_b,
                                   te1_w, te1_b, te2_w, te2_b, spa, tem);
  k_build_h<<<896, 256, 0, stream>>>(x, bn0, je1_w, je1_b, je2_w, je2_b, spa, H1);
  k_attn<<<896, 256, SHM_ATTN, stream>>>(H1, g1_w, g1_b, g2_w, g2_b, G);
  k_softmax<<<25088, 256, 0, stream>>>(G, Gbf);
  k_gcn<128, 128><<<896, 256, SHM_G128, stream>>>(H1, Gbf, gcn1_w, gcn1_w1, gcn1_b, gcn1_bn, BA);
  k_gcn<128, 256><<<896, 256, SHM_G128, stream>>>(BA, Gbf, gcn2_w, gcn2_w1, gcn2_b, gcn2_bn, BB);
  k_gcn<256, 256><<<896, 256, SHM_G256, stream>>>(BB, Gbf, gcn3_w, gcn3_w1, gcn3_b, gcn3_bn, BA);
  k_post<<<896, 256, 0, stream>>>(BA, tem, hmax);
  k_final<<<4, 256, 0, stream>>>(hmax, loc1_w, loc1_b, loc1_bn,
                                 loc2_w, loc2_b, loc2_bn, (float*)d_out);
}
